// LSTMMLP_PSO_28252294873520
// MI455X (gfx1250) — matrix operand order repaired and checked by reading, not compiled, not run on hardware
//
#include <hip/hip_runtime.h>
#include <hip/hip_bf16.h>
#include <cstdint>

typedef float        v2f    __attribute__((ext_vector_type(2)));
typedef float        v8f    __attribute__((ext_vector_type(8)));
typedef __bf16       v16bf  __attribute__((ext_vector_type(16)));
typedef unsigned int v8u    __attribute__((ext_vector_type(8)));

#define T_STEPS 60
#define FIN 5
#define HID 64
#define NZ 256          // 4*H
#define MLP_D1 64
#define MLP_D2 32
#define CDIM 65         // H + S
#define BN_EPS 1e-3f

#define ROWS_PER_WAVE  16
#define ROWS_PER_BLK   64
#define BLK_THREADS    128

#if __has_builtin(__builtin_amdgcn_wmma_f32_16x16x4_f32)
#define USE_F32WMMA 1
#else
#define USE_F32WMMA 0
#endif

// i/f/o gate rows of U/W/b are pre-scaled by 0.5 at setup (exact: exponent decrement),
// so their WMMA output is already z/2 and sigmoid(z) = 0.5*tanh(z/2)+0.5 needs no mul.
#if __has_builtin(__builtin_amdgcn_tanhf)
__device__ __forceinline__ float tanhx(float x) { return __builtin_amdgcn_tanhf(x); }
__device__ __forceinline__ float sig_from_half(float zh) {  // zh = z/2
  return fmaf(0.5f, __builtin_amdgcn_tanhf(zh), 0.5f);
}
#else
__device__ __forceinline__ float tanhx(float x) {
  return 1.0f - 2.0f * __builtin_amdgcn_rcpf(1.0f + __expf(2.0f * x));
}
__device__ __forceinline__ float sig_from_half(float zh) {  // sigmoid(2*zh)
  return __builtin_amdgcn_rcpf(1.0f + __expf(-2.0f * zh));
}
#endif

__device__ __forceinline__ unsigned int packbf(float a, float b) {
#if __has_builtin(__builtin_amdgcn_cvt_pk_bf16_f32)
  return __builtin_bit_cast(unsigned int, __builtin_amdgcn_cvt_pk_bf16_f32(a, b));
#else
  unsigned short lo = __builtin_bit_cast(unsigned short, (__bf16)a);
  unsigned short hi = __builtin_bit_cast(unsigned short, (__bf16)b);
  return (unsigned int)lo | ((unsigned int)hi << 16);
#endif
}

// K-slot permutation: WMMA k-slot kl (0..31) of fragment kc maps to H-dim d such that
// the lane-half that must supply B slot kl holds that dim locally in D layout.
__device__ __forceinline__ int kslot_dim(int kc, int kl) {
  return 32 * kc + kl;
}

__global__ __launch_bounds__(BLK_THREADS)
void lstm_mlp_fused(const float* __restrict__ x_seq, const float* __restrict__ x_static,
                    const float* __restrict__ W,  const float* __restrict__ U,  const float* __restrict__ b,
                    const float* __restrict__ bn1_g, const float* __restrict__ bn1_b,
                    const float* __restrict__ bn1_m, const float* __restrict__ bn1_v,
                    const float* __restrict__ bn2_g, const float* __restrict__ bn2_b,
                    const float* __restrict__ bn2_m, const float* __restrict__ bn2_v,
                    const float* __restrict__ W1, const float* __restrict__ b1,
                    const float* __restrict__ W2, const float* __restrict__ b2,
                    const float* __restrict__ Wo, const float* __restrict__ bo,
                    float* __restrict__ out)
{
  // U^T pre-swizzled into bf16 WMMA A-fragment order (K permuted by kslot_dim)
  __shared__ __align__(32) __bf16 Ulds[32 * 512];          // 32 KB
  // W' = [W rows 0..4 ; b ; 0 ; 0], plain [M][K] f32
  __shared__ __align__(16) float  Wlds[NZ * 8];            // 8 KB
  __shared__ float hstage[ROWS_PER_BLK][CDIM + 1];         // 16.9 KB

  const int tid  = threadIdx.x;
  const int lane = tid & 31;
  const int wave = tid >> 5;
  const int n    = lane & 15;           // batch column within the wave tile
  const bool hiHalf = (lane & 16) != 0;

  // ---------------- one-time setup: swizzle U^T into A-fragment order ----------------
  for (int idx = tid; idx < 32 * 512; idx += BLK_THREADS) {
    int frag = idx >> 9, within = idx & 511;
    int l = within >> 4, j = within & 15;
    int v = j >> 1, hl = j & 1;
    int mt = frag >> 1, kc = frag & 1;
    int koff = (v < 4 ? 2 * v : 2 * v + 8) + ((l & 16) ? 8 : 0) + hl;  // A-slot -> k
    int M = mt * 16 + (l & 15);
    int Kh = kslot_dim(kc, koff);                                      // k -> H dim (permuted)
    float uval = U[Kh * NZ + M];          // U is (H,4H) row-major; U^T[M][Kh] = U[Kh][M]
    if (M < 128 || M >= 192) uval *= 0.5f;  // i,f,o rows -> sigmoid gets z/2 for free
    Ulds[idx] = (__bf16)uval;
  }
  for (int idx = tid; idx < NZ * 8; idx += BLK_THREADS) {
    int m = idx >> 3, k = idx & 7;
    float val = (k < FIN) ? W[k * NZ + m] : (k == FIN ? b[m] : 0.0f);
    if (m < 128 || m >= 192) val *= 0.5f;   // same 0.5 pre-scale for i,f,o rows
    Wlds[idx] = val;
  }
  __syncthreads();

  // ---------------- LSTM recurrence (transposed, 16 batch rows per wave) ----------------
  const int row0 = blockIdx.x * ROWS_PER_BLK + wave * ROWS_PER_WAVE;
  const float* xp = x_seq + (size_t)(row0 + n) * T_STEPS * FIN;

  float cst[4][8];    // c state, D layout: chunk hc covers H dims hc*16..+15
  float hkeep[4][8];  // masked h (fp32, D layout) — feeds next-step hB and the MLP epilogue
  v8u hB[2] = { {0,0,0,0,0,0,0,0}, {0,0,0,0,0,0,0,0} };
#pragma unroll
  for (int hc = 0; hc < 4; ++hc)
#pragma unroll
    for (int r = 0; r < 8; ++r) { cst[hc][r] = 0.f; hkeep[hc][r] = 0.f; }

#pragma unroll 1
  for (int t = 0; t < T_STEPS; ++t) {
    float x0 = xp[0], x1 = xp[1], x2 = xp[2], x3 = xp[3], x4 = xp[4];
    xp += FIN;
    // mask = any(x != 0) as a {0,1} float (abs modifiers are free on VALU)
    float sabs = fabsf(x0) + fabsf(x1) + fabsf(x2) + fabsf(x3) + fabsf(x4);
    float mf = (sabs != 0.0f) ? 1.0f : 0.0f;

#if USE_F32WMMA
    // x^T_aug as f32 B operands (4x16): lanes0-15 hold K={0,1}/{4,5}, lanes16-31 K={2,3}/{6,7}
    v2f xb0, xb1;
    xb0.x = hiHalf ? x2 : x0;   xb0.y = hiHalf ? x3 : x1;
    xb1.x = hiHalf ? 0.f : x4;  xb1.y = hiHalf ? 0.f : 1.0f;   // constant-1 feature carries bias b
#endif

    v16bf hb0 = __builtin_bit_cast(v16bf, hB[0]);
    v16bf hb1 = __builtin_bit_cast(v16bf, hB[1]);

#pragma unroll
    for (int hc = 0; hc < 4; ++hc) {
      v8f acc[4];
#pragma unroll
      for (int gI = 0; gI < 4; ++gI) {        // i, f, g, o (i/f/o rows pre-scaled by 0.5)
        const int mt = gI * 4 + hc;           // z row-tile
        v8f a = {0.f,0.f,0.f,0.f,0.f,0.f,0.f,0.f};
#if USE_F32WMMA
        {
          const int M8 = (mt * 16 + n) * 8 + (hiHalf ? 2 : 0);
          v2f wa0 = *(const v2f*)&Wlds[M8];
          v2f wa1 = *(const v2f*)&Wlds[M8 + 4];
          a = __builtin_amdgcn_wmma_f32_16x16x4_f32(false, wa0, false, xb0, (short)0, a, false, false);
          a = __builtin_amdgcn_wmma_f32_16x16x4_f32(false, wa1, false, xb1, (short)0, a, false, false);
        }
#else
        {
          float xa[6] = {x0, x1, x2, x3, x4, 1.0f};
#pragma unroll
          for (int r = 0; r < 8; ++r) {
            int Mr = mt * 16 + r + (hiHalf ? 8 : 0);
            float s = 0.f;
#pragma unroll
            for (int k = 0; k < 6; ++k) s += Wlds[Mr * 8 + k] * xa[k];
            a[r] = s;
          }
        }
#endif
        v16bf ua0 = *(const v16bf*)&Ulds[(mt * 2 + 0) * 512 + lane * 16];
        v16bf ua1 = *(const v16bf*)&Ulds[(mt * 2 + 1) * 512 + lane * 16];
        a = __builtin_amdgcn_wmma_f32_16x16x32_bf16(false, ua0, false, hb0, (short)0, a, false, false);
        a = __builtin_amdgcn_wmma_f32_16x16x32_bf16(false, ua1, false, hb1, (short)0, a, false, false);
        acc[gI] = a;
      }
      // gate math, elementwise per lane; branchless masked update via fmaf blend
#pragma unroll
      for (int r = 0; r < 8; ++r) {
        float iv = sig_from_half(acc[0][r]);   // acc already holds z/2 for i,f,o
        float fv = sig_from_half(acc[1][r]);
        float gv = tanhx(acc[2][r]);           // g rows unscaled
        float ov = sig_from_half(acc[3][r]);
        float cn = fv * cst[hc][r] + iv * gv;
        float hn = ov * tanhx(cn);
        cst[hc][r]   = fmaf(mf, cn - cst[hc][r],   cst[hc][r]);
        hkeep[hc][r] = fmaf(mf, hn - hkeep[hc][r], hkeep[hc][r]);
      }
    }

    // rebuild h^T bf16 B fragments: thanks to the K permutation, every B slot is supplied
    // by a locally-held hkeep register — identical code for both lane halves, no shuffles.
#pragma unroll
    for (int kc = 0; kc < 2; ++kc) {
      v8u packed;
#pragma unroll
      for (int v = 0; v < 4; ++v)
        packed[v] = packbf(hkeep[2 * kc][2 * v], hkeep[2 * kc][2 * v + 1]);
#pragma unroll
      for (int v = 4; v < 8; ++v)
        packed[v] = packbf(hkeep[2 * kc + 1][2 * v - 8], hkeep[2 * kc + 1][2 * v - 7]);
      hB[kc] = packed;
    }
  }

  // ---------------- stage h to LDS (row-major per batch row) ----------------
#pragma unroll
  for (int hc = 0; hc < 4; ++hc)
#pragma unroll
    for (int r = 0; r < 8; ++r) {
      int m = hc * 16 + r + (hiHalf ? 8 : 0);
      hstage[wave * 16 + n][m] = hkeep[hc][r];
    }
  __syncthreads();

  // ---------------- BN1 -> concat -> BN2 -> MLP (one batch row per thread) ----------------
  if (tid < ROWS_PER_BLK) {
    const int gr = blockIdx.x * ROWS_PER_BLK + tid;
#pragma unroll 1
    for (int m = 0; m < HID; ++m) {
      float h = hstage[tid][m];
      h = (h - bn1_m[m]) * (bn1_g[m] * rsqrtf(bn1_v[m] + BN_EPS)) + bn1_b[m];
      h = (h - bn2_m[m]) * (bn2_g[m] * rsqrtf(bn2_v[m] + BN_EPS)) + bn2_b[m];
      hstage[tid][m] = h;
    }
    {
      float h = x_static[gr];
      h = (h - bn2_m[HID]) * (bn2_g[HID] * rsqrtf(bn2_v[HID] + BN_EPS)) + bn2_b[HID];
      hstage[tid][HID] = h;
    }
    float a2[MLP_D2];
#pragma unroll
    for (int e = 0; e < MLP_D2; ++e) a2[e] = b2[e];
#pragma unroll 1
    for (int d = 0; d < MLP_D1; ++d) {
      float s = b1[d];
#pragma unroll 5
      for (int c = 0; c < CDIM; ++c) s += hstage[tid][c] * W1[c * MLP_D1 + d];
      s = fmaxf(s, 0.f);
#pragma unroll
      for (int e = 0; e < MLP_D2; ++e) a2[e] += s * W2[d * MLP_D2 + e];
    }
    float o = bo[0];
#pragma unroll
    for (int e = 0; e < MLP_D2; ++e) o += fmaxf(a2[e], 0.f) * Wo[e];
    out[gr] = o;
  }
}

extern "C" void kernel_launch(void* const* d_in, const int* in_sizes, int n_in,
                              void* d_out, int out_size, void* d_ws, size_t ws_size,
                              hipStream_t stream) {
  const float* x_seq    = (const float*)d_in[0];
  const float* x_static = (const float*)d_in[1];
  const float* W        = (const float*)d_in[2];
  const float* U        = (const float*)d_in[3];
  const float* b        = (const float*)d_in[4];
  const float* bn1_g    = (const float*)d_in[5];
  const float* bn1_b    = (const float*)d_in[6];
  const float* bn1_m    = (const float*)d_in[7];
  const float* bn1_v    = (const float*)d_in[8];
  const float* bn2_g    = (const float*)d_in[9];
  const float* bn2_b    = (const float*)d_in[10];
  const float* bn2_m    = (const float*)d_in[11];
  const float* bn2_v    = (const float*)d_in[12];
  const float* W1       = (const float*)d_in[13];
  const float* b1       = (const float*)d_in[14];
  const float* W2       = (const float*)d_in[15];
  const float* b2       = (const float*)d_in[16];
  const float* Wo       = (const float*)d_in[17];
  const float* bo       = (const float*)d_in[18];
  float* out = (float*)d_out;

  const int grid = (out_size + ROWS_PER_BLK - 1) / ROWS_PER_BLK;  // 16384/64 = 256
  lstm_mlp_fused<<<grid, BLK_THREADS, 0, stream>>>(
      x_seq, x_static, W, U, b,
      bn1_g, bn1_b, bn1_m, bn1_v,
      bn2_g, bn2_b, bn2_m, bn2_v,
      W1, b1, W2, b2, Wo, bo, out);
}
